// HamiHead_84748294684830
// MI455X (gfx1250) — matrix operand order repaired and checked by reading, not compiled, not run on hardware
//
#include <hip/hip_runtime.h>
#include <hip/hip_bf16.h>

typedef __attribute__((ext_vector_type(16))) _Float16 v16h;
typedef __attribute__((ext_vector_type(4)))  _Float16 v4h;
typedef __attribute__((ext_vector_type(8)))  float    v8f;

#define N_NODES 2560
#define N_EDGES 48640
#define EDIM 800
#define NPW 1792
#define HS 128

// ---------------- instruction / group metadata (e3nn Expansion) ----------------
__constant__ signed char INS_LI[19]={0,0,0,1,1,1,1,1,1,2,2,2,2,2,2,3,3,3,4};
__constant__ signed char INS_L1[19]={0,1,2,0,1,1,1,2,2,0,1,1,2,2,2,1,2,2,2};
__constant__ signed char INS_L2[19]={0,1,2,1,0,1,2,1,2,2,1,2,0,1,2,2,1,2,2};
__constant__ short INS_XOFF[19]={0,0,0,32,32,32,32,32,32,128,128,128,128,128,128,288,288,288,512};
__constant__ short INS_WI[19]={0,288,416,448,640,832,960,1024,1088,1120,1216,1344,1408,1504,1568,1600,1664,1728,1760};
__constant__ signed char INS_BI[19]={0,9,13,-1,-1,-1,-1,-1,-1,-1,-1,-1,-1,-1,-1,-1,-1,-1,-1};
__constant__ short INS_W3[19]={0,1,10,35,44,53,80,125,170,245,270,315,390,415,490,615,720,825,1000};

// 36 output cell-groups (j,k,u,v) tiling the 14x14 block
__constant__ signed char G_J[36]={0,0,0,0,0,0,0,0,0, 0,0,0,0,0,0, 0,0,0, 1,1,1,1,1,1, 1,1,1,1, 1,1, 2,2,2, 2,2, 2};
__constant__ signed char G_K[36]={0,0,0,0,0,0,0,0,0, 1,1,1,1,1,1, 2,2,2, 0,0,0,0,0,0, 1,1,1,1, 2,2, 0,0,0, 1,1, 2};
__constant__ signed char G_U[36]={0,0,0,1,1,1,2,2,2, 0,0,1,1,2,2, 0,1,2, 0,0,0,1,1,1, 0,0,1,1, 0,1, 0,0,0, 0,0, 0};
__constant__ signed char G_V[36]={0,1,2,0,1,2,0,1,2, 0,1,0,1,0,1, 0,0,0, 0,1,2,0,1,2, 0,1,0,1, 0,0, 0,1,2, 0,1, 0};
__constant__ signed char G_N[36]={1,1,1,1,1,1,1,1,1, 1,1,1,1,1,1, 1,1,1, 1,1,1,1,1,1, 3,3,3,3, 3,3, 1,1,1, 3,3, 5};
__constant__ short G_S[36]={0,1,2,3,4,5,6,7,8, 9,10,11,12,13,14, 15,16,17, 18,19,20,21,22,23, 24,27,30,33, 36,39, 42,43,44, 45,48, 51};
__constant__ signed char G_FLAT[56]={0,0,0,0,0,0,0,0,0, 3,3,3,3,3,3, 9,9,9, 4,4,4,4,4,4,
                                     1,5,10,1,5,10,1,5,10,1,5,10, 6,11,15,6,11,15,
                                     12,12,12, 7,13,16,7,13,16, 2,8,14,17,18};
__constant__ signed char RB3[3]={0,3,9};

// ---------------- Wigner-3j on-device precompute (fp64) ----------------
struct C2 { double re, im; };

__device__ double dfact(int n){ double r=1.0; for(int i=2;i<=n;++i) r*=i; return r; }

__device__ double su2_cg(int j1,int m1,int j2,int m2,int j3,int m3){
  if (m1+m2 != m3) return 0.0;
  int vmin = max(max(-j1+j2+m3, -j1+m1), 0);
  int vmax = min(min(j2+j3+m1, j3-j1+j2), j3+m3);
  double C = sqrt((double)(2*j3+1) *
      dfact(j3+j1-j2)*dfact(j3-j1+j2)*dfact(j1+j2-j3)*dfact(j3+m3)*dfact(j3-m3) /
      (dfact(j1+j2+j3+1)*dfact(j1-m1)*dfact(j1+m1)*dfact(j2-m2)*dfact(j2+m2)));
  double S = 0.0;
  for (int v=vmin; v<=vmax; ++v){
    double term = dfact(j2+j3+m1-v)*dfact(j1-m1+v) /
      (dfact(v)*dfact(j3-j1+j2-v)*dfact(j3+m3-v)*dfact(v+j1-j2-m3));
    S += ((v+j2+m2)&1) ? -term : term;
  }
  return C*S;
}

__device__ void build_q(int l, C2 q[9][9]){
  for (int a=0;a<9;++a) for (int b=0;b<9;++b){ q[a][b].re=0.0; q[a][b].im=0.0; }
  double s = 0.70710678118654752440;
  for (int m=-l;m<0;++m){ q[l+m][l-m].re = s; q[l+m][l+m].im = -s; }
  q[l][l].re = 1.0;
  for (int m=1;m<=l;++m){ double sg=(m&1)?-1.0:1.0; q[l+m][l+m].re=sg*s; q[l+m][l-m].im=sg*s; }
  double fr,fi;
  switch(l&3){ case 0: fr=1;fi=0;break; case 1: fr=0;fi=-1;break; case 2: fr=-1;fi=0;break; default: fr=0;fi=1;break; }
  for (int a=0;a<2*l+1;++a) for (int b=0;b<2*l+1;++b){
    double re=q[a][b].re, im=q[a][b].im;
    q[a][b].re = re*fr - im*fi;
    q[a][b].im = re*fi + im*fr;
  }
}

__global__ void __launch_bounds__(32) k_w3j(float* w3g){
  if (threadIdx.x != 0) return;
  int t = blockIdx.x;
  int l1 = INS_L1[t], l2 = INS_L2[t], l3 = INS_LI[t];
  int d1 = 2*l1+1, d2 = 2*l2+1, d3 = 2*l3+1;
  C2 q1[9][9], q2[9][9], q3[9][9];
  build_q(l1,q1); build_q(l2,q2); build_q(l3,q3);
  double CG[5][5][9];
  for (int a=0;a<5;++a) for(int b=0;b<5;++b) for(int c=0;c<9;++c) CG[a][b][c]=0.0;
  for (int i=0;i<d1;++i) for (int k=0;k<d2;++k){
    int m1=i-l1, m2=k-l2, m3=m1+m2;
    if (m3>=-l3 && m3<=l3) CG[i][k][m3+l3] = su2_cg(l1,m1,l2,m2,l3,m3);
  }
  double outv[5][5][9]; double nrm=0.0;
  for (int jj=0;jj<d1;++jj) for (int ll=0;ll<d2;++ll) for (int mm=0;mm<d3;++mm){
    double ar=0.0;
    for (int i=0;i<d1;++i) for (int k=0;k<d2;++k){
      double cr = q1[i][jj].re*q2[k][ll].re - q1[i][jj].im*q2[k][ll].im;
      double ci = q1[i][jj].re*q2[k][ll].im + q1[i][jj].im*q2[k][ll].re;
      if (cr==0.0 && ci==0.0) continue;
      for (int n=0;n<d3;++n){
        double cg = CG[i][k][n];
        if (cg==0.0) continue;
        // multiply (cr,ci) by conj(q3[n][mm]) and take real part
        ar += cg*(cr*q3[n][mm].re + ci*q3[n][mm].im);
      }
    }
    outv[jj][ll][mm]=ar; nrm += ar*ar;
  }
  double inv = 1.0/sqrt(nrm);
  float* dst = w3g + INS_W3[t];
  for (int jj=0;jj<d1;++jj) for (int ll=0;ll<d2;++ll) for (int mm=0;mm<d3;++mm)
    dst[(jj*d2+ll)*d3+mm] = (float)(outv[jj][ll][mm]*inv);
}

// ---------------- pack f32 weights [fi,fo] into WMMA B-fragment f16 layout ----------------
// layout: [fo/16][fi/32][lane(32)][e(16)] ; value = W[kb*32 + (lane/16)*8 + (e%8) + (e/8)*16][t*16 + lane%16]
__global__ void k_pack(const float* __restrict__ src, _Float16* __restrict__ dst, int fi, int fo){
  int i = blockIdx.x*256 + threadIdx.x;
  int total = fi*fo;
  if (i >= total) return;
  int KB = fi >> 5;
  int e = i & 15;
  int lane = (i >> 4) & 31;
  int kb = (i >> 9) % KB;
  int t  = i / (KB << 9);
  int k = kb*32 + (lane>>4)*8 + (e&7) + (e>>3)*16;
  int n = t*16 + (lane&15);
  dst[i] = (_Float16)src[(size_t)k*fo + n];
}

// ---------------- layer-1 dual GEMM + SiLU -> two f16 hidden activations ----------------
// 16 samples x 128 cols per block; 8 waves, one 16-col tile per wave; one shared A
// fragment feeds two WMMAs (main-MLP weights and bias-MLP weights).
__global__ void __launch_bounds__(256) k_l1(const float* __restrict__ attr,
                                            const int* __restrict__ dstI, const int* __restrict__ srcI,
                                            const _Float16* __restrict__ BpA, const float* __restrict__ biasA,
                                            _Float16* __restrict__ HoutA,
                                            const _Float16* __restrict__ BpB, const float* __restrict__ biasB,
                                            _Float16* __restrict__ HoutB, int kin){
  int s0 = blockIdx.x * 16;
  int wave = threadIdx.x >> 5, lane = threadIdx.x & 31;
  int m = lane & 15, half = lane >> 4;
  int KB = kin >> 5;
  int samp = s0 + m;
  int rd = 0, rs = 0;
  if (dstI){ rd = dstI[samp]; rs = srcI[samp]; }
  v8f accA = {};
  v8f accB = {};
  for (int kb=0; kb<KB; ++kb){
    union { v16h v; _Float16 h[16]; } af;
    int c0 = kb*32 + half*8;
    const float* rp;
    if (dstI){
      if (c0 < HS) rp = attr + (size_t)rd*HS + c0;
      else         rp = attr + (size_t)rs*HS + (c0 - HS);
    } else         rp = attr + (size_t)samp*HS + c0;
    float4 f0 = *(const float4*)rp;
    float4 f1 = *(const float4*)(rp+4);
    float4 f2 = *(const float4*)(rp+16);
    float4 f3 = *(const float4*)(rp+20);
    af.h[0]=(_Float16)f0.x; af.h[1]=(_Float16)f0.y; af.h[2]=(_Float16)f0.z; af.h[3]=(_Float16)f0.w;
    af.h[4]=(_Float16)f1.x; af.h[5]=(_Float16)f1.y; af.h[6]=(_Float16)f1.z; af.h[7]=(_Float16)f1.w;
    af.h[8]=(_Float16)f2.x; af.h[9]=(_Float16)f2.y; af.h[10]=(_Float16)f2.z; af.h[11]=(_Float16)f2.w;
    af.h[12]=(_Float16)f3.x; af.h[13]=(_Float16)f3.y; af.h[14]=(_Float16)f3.z; af.h[15]=(_Float16)f3.w;
    size_t boff = ((size_t)(wave*KB + kb)*32 + lane)*16;
    union { v16h v; uint4 q[2]; } bfA;
    const uint4* bpA = (const uint4*)(BpA + boff);
    bfA.q[0] = bpA[0]; bfA.q[1] = bpA[1];
    accA = __builtin_amdgcn_wmma_f32_16x16x32_f16(false, af.v, false, bfA.v, (short)0, accA, false, false);
    union { v16h v; uint4 q[2]; } bfB;
    const uint4* bpB = (const uint4*)(BpB + boff);
    bfB.q[0] = bpB[0]; bfB.q[1] = bpB[1];
    accB = __builtin_amdgcn_wmma_f32_16x16x32_f16(false, af.v, false, bfB.v, (short)0, accB, false, false);
  }
  int col = wave*16 + m;
  float bvA = biasA[col];
  float bvB = biasB[col];
  #pragma unroll
  for (int r=0;r<8;++r){
    size_t row = (size_t)(s0 + r + 8*half)*HS + col;
    float xA = accA[r] + bvA;
    HoutA[row] = (_Float16)(xA / (1.0f + __expf(-xA)));
    float xB = accB[r] + bvB;
    HoutB[row] = (_Float16)(xB / (1.0f + __expf(-xB)));
  }
}

// ---------------- layer-2 GEMM (WMMA, fused) + e3nn Expansion ----------------
// 32 samples per block, 512 threads = 16 waves = 8 column groups x 2 row halves.
// W tile (32x1792 f32) + x tile (32x800 f16) + bias + W3J live in LDS (~287 KB).
__global__ void __launch_bounds__(512) k2(
  const float* __restrict__ fii, const float* __restrict__ fij,
  const _Float16* __restrict__ h_ii, const _Float16* __restrict__ hb_ii,
  const _Float16* __restrict__ h_ij, const _Float16* __restrict__ hb_ij,
  const _Float16* __restrict__ p2_ii, const _Float16* __restrict__ p2_ij,
  const float* __restrict__ bii2, const float* __restrict__ bij2,
  const float* __restrict__ Wiib2, const float* __restrict__ biib2,
  const float* __restrict__ Wijb2, const float* __restrict__ bijb2,
  const float* __restrict__ w3g, float* __restrict__ outDiag, float* __restrict__ outEdge)
{
  extern __shared__ __align__(16) char smem[];
  float*    Wl  = (float*)smem;                        // 32*1792 f32 = 229376 B
  float*    bl  = (float*)(smem + 229376);             // 32*14 f32   = 1792 B
  float*    w3l = (float*)(smem + 229376 + 1792);      // 1225 f32 (pad 4992 B)
  _Float16* xl  = (_Float16*)(smem + 229376 + 1792 + 4992); // 32*800 f16 = 51200 B

  int bt = blockIdx.x;
  bool diag = bt < (N_NODES/32);
  int s0 = diag ? bt*32 : (bt - N_NODES/32)*32;
  const float*    xsrc = diag ? fii   : fij;
  const _Float16* h    = diag ? h_ii  : h_ij;
  const _Float16* hb   = diag ? hb_ii : hb_ij;
  const _Float16* p2   = diag ? p2_ii : p2_ij;
  const float*    b2   = diag ? bii2  : bij2;
  const float*    W2b  = diag ? Wiib2 : Wijb2;
  const float*    b2b  = diag ? biib2 : bijb2;
  float* outBase = (diag ? outDiag : outEdge) + (size_t)s0*196;
  int tid = threadIdx.x;

  // load x tile (f32 -> f16)
  const float4* xs4 = (const float4*)(xsrc + (size_t)s0*EDIM);
  for (int i=tid; i<32*EDIM/4; i+=512){
    float4 f = xs4[i];
    v4h hh = {(_Float16)f.x,(_Float16)f.y,(_Float16)f.z,(_Float16)f.w};
    ((v4h*)xl)[i] = hh;
  }
  // copy W3J tables
  for (int i=tid; i<1225; i+=512) w3l[i] = w3g[i];
  // expansion bias vector: silu_hb @ W2b + b2b  (32 samples x 14)
  if (tid < 448){
    int s = tid/14, c = tid%14;
    const _Float16* hr = hb + (size_t)(s0+s)*HS;
    float a = b2b[c];
    for (int hh=0; hh<HS; ++hh) a += (float)hr[hh] * W2b[hh*14 + c];
    bl[s*14+c] = a;
  }

  // WMMA: [32,128] x [128,1792] -> LDS (f32)
  int wave = tid>>5, lane = tid&31, m = lane&15, half = lane>>4;
  int colGroup = wave & 7;       // 8 groups x 14 tiles = 112 column tiles
  int rowHalf  = wave >> 3;      // 0: samples 0-15, 1: samples 16-31
  v8f zero = {};
  v8f acc[14];
  #pragma unroll
  for (int t=0;t<14;++t) acc[t] = zero;
  const _Float16* hrow = h + (size_t)(s0 + rowHalf*16 + m)*HS;
  for (int kb=0; kb<4; ++kb){
    union { v16h v; uint4 q[2]; } af;
    af.q[0] = *(const uint4*)(hrow + kb*32 + half*8);
    af.q[1] = *(const uint4*)(hrow + kb*32 + half*8 + 16);
    #pragma unroll
    for (int t=0; t<14; ++t){
      int tg = colGroup*14 + t;
      union { v16h v; uint4 q[2]; } bf;
      const uint4* bp = (const uint4*)(p2 + ((size_t)(tg*4 + kb)*32 + lane)*16);
      bf.q[0] = bp[0]; bf.q[1] = bp[1];
      acc[t] = __builtin_amdgcn_wmma_f32_16x16x32_f16(false, af.v, false, bf.v, (short)0, acc[t], false, false);
    }
  }
  #pragma unroll
  for (int t=0; t<14; ++t){
    int colg = (colGroup*14+t)*16 + m;
    float bv = b2[colg];
    #pragma unroll
    for (int r=0;r<8;++r)
      Wl[(size_t)(rowHalf*16 + r + 8*half)*NPW + colg] = acc[t][r] + bv;
  }
  __syncthreads();

  // Expansion: 16 threads per sample, each owns 2-3 disjoint (j,k,u,v) cell groups
  int s = tid >> 4, l16 = tid & 15;
  for (int g=l16; g<36; g+=16){
    int j = G_J[g], k = G_K[g];
    int d1 = 2*j+1, d2 = 2*k+1;
    int mul2 = 3-k;
    int wstride = (3-j)*mul2;
    int u = G_U[g], v = G_V[g];
    float accq[25];
    for (int z=0; z<25; ++z) accq[z]=0.f;
    int ns = G_N[g], st = G_S[g];
    for (int ii=0; ii<ns; ++ii){
      int t = G_FLAT[st+ii];
      int li = INS_LI[t], dk = 2*li+1;
      float res[9];
      for (int z=0; z<9; ++z) res[z]=0.f;
      const _Float16* xr = xl + s*EDIM + INS_XOFF[t];
      const float* wr = Wl + s*NPW + INS_WI[t] + u*mul2 + v;
      for (int w=0; w<32; ++w){
        float wv = wr[w*wstride];
        const _Float16* xp = xr + w*dk;
        for (int km=0; km<dk; ++km) res[km] += wv * (float)xp[km];
      }
      if (INS_BI[t] >= 0) res[0] += bl[s*14 + INS_BI[t] + u*mul2 + v];
      const float* C3 = w3l + INS_W3[t];
      for (int a=0; a<d1; ++a)
        for (int b=0; b<d2; ++b){
          const float* cp = C3 + (a*d2+b)*dk;
          float sum=0.f;
          for (int km=0; km<dk; ++km) sum += cp[km]*res[km];
          accq[a*d2+b] += sum;
        }
    }
    int ro = RB3[j] + u*d1, co = RB3[k] + v*d2;
    float* ob = outBase + (size_t)s*196;
    for (int a=0; a<d1; ++a)
      for (int b=0; b<d2; ++b)
        ob[(ro+a)*14 + (co+b)] = accq[a*d2+b]*0.03125f;
  }
}

// ---------------- in-place symmetrization ----------------
__global__ void k_sym_diag(float* out){
  int idx = blockIdx.x*256 + threadIdx.x;
  if (idx >= N_NODES*196) return;
  int n = idx/196, rc = idx%196, r = rc/14, c = rc%14;
  if (r > c) return;
  float* b = out + (size_t)n*196;
  float x = b[r*14+c], y = b[c*14+r];
  float vv = 0.5f*(x+y);
  b[r*14+c] = vv; b[c*14+r] = vv;
}

__global__ void k_sym_edge(float* out, const int* __restrict__ tidx){
  int idx = blockIdx.x*256 + threadIdx.x;
  if (idx >= N_EDGES*196) return;
  int e = idx/196, rc = idx%196, r = rc/14, c = rc%14;
  int te = tidx[e];
  if (e >= te) return;
  float* be = out + (size_t)e*196;
  float* bt = out + (size_t)te*196;
  float x = be[r*14+c], y = bt[c*14+r];
  float vv = 0.5f*(x+y);
  be[r*14+c] = vv; bt[c*14+r] = vv;
}

extern "C" void kernel_launch(void* const* d_in, const int* in_sizes, int n_in,
                              void* d_out, int out_size, void* d_ws, size_t ws_size,
                              hipStream_t stream){
  (void)in_sizes; (void)n_in; (void)out_size; (void)ws_size;
  const float* fii   = (const float*)d_in[0];
  const float* fij   = (const float*)d_in[1];
  const float* attr  = (const float*)d_in[2];
  const float* Wii1  = (const float*)d_in[3];  const float* bii1  = (const float*)d_in[4];
  const float* Wii2  = (const float*)d_in[5];  const float* bii2  = (const float*)d_in[6];
  const float* Wiib1 = (const float*)d_in[7];  const float* biib1 = (const float*)d_in[8];
  const float* Wiib2 = (const float*)d_in[9];  const float* biib2 = (const float*)d_in[10];
  const float* Wij1  = (const float*)d_in[11]; const float* bij1  = (const float*)d_in[12];
  const float* Wij2  = (const float*)d_in[13]; const float* bij2  = (const float*)d_in[14];
  const float* Wijb1 = (const float*)d_in[15]; const float* bijb1 = (const float*)d_in[16];
  const float* Wijb2 = (const float*)d_in[17]; const float* bijb2 = (const float*)d_in[18];
  const int* fei = (const int*)d_in[19];
  const int* tei = (const int*)d_in[20];
  const int* dstI = fei; const int* srcI = fei + N_EDGES;

  char* ws = (char*)d_ws;
  size_t off = 0;
  auto carve = [&](size_t bytes)->char*{ char* p = ws + off; off = (off + bytes + 255) & ~(size_t)255; return p; };
  float*    w3g    = (float*)   carve(1232*4);
  _Float16* p_ii1  = (_Float16*)carve((size_t)128*128*2);
  _Float16* p_iib1 = (_Float16*)carve((size_t)128*128*2);
  _Float16* p_ij1  = (_Float16*)carve((size_t)256*128*2);
  _Float16* p_ijb1 = (_Float16*)carve((size_t)256*128*2);
  _Float16* p_ii2  = (_Float16*)carve((size_t)128*NPW*2);
  _Float16* p_ij2  = (_Float16*)carve((size_t)128*NPW*2);
  _Float16* h_ii   = (_Float16*)carve((size_t)N_NODES*HS*2);
  _Float16* hb_ii  = (_Float16*)carve((size_t)N_NODES*HS*2);
  _Float16* h_ij   = (_Float16*)carve((size_t)N_EDGES*HS*2);
  _Float16* hb_ij  = (_Float16*)carve((size_t)N_EDGES*HS*2);

  k_w3j<<<19, 32, 0, stream>>>(w3g);
  k_pack<<<(128*128+255)/256,256,0,stream>>>(Wii1,  p_ii1,  128, 128);
  k_pack<<<(128*128+255)/256,256,0,stream>>>(Wiib1, p_iib1, 128, 128);
  k_pack<<<(256*128+255)/256,256,0,stream>>>(Wij1,  p_ij1,  256, 128);
  k_pack<<<(256*128+255)/256,256,0,stream>>>(Wijb1, p_ijb1, 256, 128);
  k_pack<<<(128*NPW+255)/256,256,0,stream>>>(Wii2,  p_ii2,  128, NPW);
  k_pack<<<(128*NPW+255)/256,256,0,stream>>>(Wij2,  p_ij2,  128, NPW);

  // fused dual layer-1 (main + bias MLP share the A gather)
  k_l1<<<N_NODES/16, 256, 0, stream>>>(attr, nullptr, nullptr,
      p_ii1, bii1, h_ii, p_iib1, biib1, hb_ii, 128);
  k_l1<<<N_EDGES/16, 256, 0, stream>>>(attr, dstI, srcI,
      p_ij1, bij1, h_ij, p_ijb1, bijb1, hb_ij, 256);

  float* outDiag = (float*)d_out;
  float* outEdge = outDiag + (size_t)N_NODES*196;
  size_t smem = 229376 + 1792 + 4992 + 51200;  // 287360 B (<320 KB per WGP workgroup)
  k2<<<(N_NODES+N_EDGES)/32, 512, smem, stream>>>(fii, fij, h_ii, hb_ii, h_ij, hb_ij,
      p_ii2, p_ij2, bii2, bij2, Wiib2, biib2, Wijb2, bijb2, w3g, outDiag, outEdge);

  k_sym_diag<<<(N_NODES*196+255)/256, 256, 0, stream>>>(outDiag);
  k_sym_edge<<<(N_EDGES*196+255)/256, 256, 0, stream>>>(outEdge, tei);
}
